// CRFLoss_82008105550085
// MI455X (gfx1250) — matrix operand order repaired and checked by reading, not compiled, not run on hardware
//
#include <hip/hip_runtime.h>

typedef _Float16 h8  __attribute__((ext_vector_type(8)));
typedef _Float16 h16 __attribute__((ext_vector_type(16)));
typedef float    f8  __attribute__((ext_vector_type(8)));

#define BB 64
#define TT 1024
#define VV 256
#define ROWS 16      // batch rows per block (one WMMA M-tile)
#define PITCH 264    // f16 row pitch: 528B = 132 dwords, 132 % 64 = 4 -> conflict-free
#define LPITCH 260   // f32 emission-buffer pitch: 8*260 % 64 = 32 -> halves on disjoint banks

__device__ __forceinline__ void async_b128_to_lds(const float* g, void* l) {
    const unsigned loff = (unsigned)(unsigned long long)(uintptr_t)l;  // LDS byte offset (low 32b)
    const unsigned long long ga = (unsigned long long)(uintptr_t)g;
    asm volatile("global_load_async_to_lds_b128 %0, %1, off"
                 :: "v"(loff), "v"(ga) : "memory");
}

__device__ __forceinline__ void wait_async0() {
#if __has_builtin(__builtin_amdgcn_s_wait_asynccnt)
    __builtin_amdgcn_s_wait_asynccnt(0);
#else
    asm volatile("s_wait_asynccnt 0x0" ::: "memory");
#endif
}

// ---------------------------------------------------------------------------
// Main forward-DP kernel: one block = 16 batch rows, whole T-scan in-kernel.
// expPT[j][i] = exp(P[i][j]) resident in LDS (f16), alpha resident in LDS (f32).
// Per step: async-stage logits[:,t,:] -> row-max -> a16=exp(alpha-m) (f16)
//           -> WMMA GEMM -> wait async -> alpha update from LDS.
// ---------------------------------------------------------------------------
__global__ __launch_bounds__(256)
void crf_forward_scan(const float* __restrict__ logits,
                      const float* __restrict__ P,
                      const float* __restrict__ S,
                      const float* __restrict__ E,
                      float* __restrict__ logZ)
{
    __shared__ __align__(16) _Float16 expPT[VV][PITCH];     // 135168 B
    __shared__ __align__(16) float    alphaS[ROWS][VV];     //  16384 B
    __shared__ __align__(16) _Float16 aexp[ROWS][PITCH];    //   8448 B
    __shared__ __align__(16) float    lbuf[ROWS][LPITCH];   //  16640 B
    __shared__ float pred[ROWS][16];
    __shared__ float rowm[ROWS];

    const int tid  = threadIdx.x;
    const int lane = tid & 31;
    const int wave = tid >> 5;          // 8 waves
    const int b0   = blockIdx.x * ROWS;

    // one-time: transposed exp(P) into LDS
    for (int idx = tid; idx < VV * VV; idx += 256) {
        const int i = idx >> 8;         // P row (source state)
        const int j = idx & 255;        // P col (dest state)
        expPT[j][i] = (_Float16)__expf(P[idx]);
    }
    // init alpha0 = S + logits[:,0,:]
    for (int idx = tid; idx < ROWS * VV; idx += 256) {
        const int r = idx >> 8;
        const int j = idx & 255;
        alphaS[r][j] = S[j] + logits[((size_t)(b0 + r) * TT) * VV + j];
    }
    __syncthreads();

    const int r8   = tid >> 4;          // 0..15 : row for reductions
    const int seg  = tid & 15;          // 16-wide segment within row
    const int mrow = lane & 15;         // WMMA: A row / B column index
    const int half = lane >> 4;         // WMMA: lane half

    for (int t = 1; t < TT; ++t) {
        // ---- issue async stage of logits[:, t, :] into lbuf (coalesced b128) ----
        #pragma unroll
        for (int i = 0; i < 4; ++i) {
            const int e  = i * 256 + tid;      // float4 element index, 1024 total
            const int r  = e >> 6;             // 64 float4 per row
            const int c4 = e & 63;
            async_b128_to_lds(&logits[((size_t)(b0 + r) * TT + t) * VV + c4 * 4],
                              &lbuf[r][c4 * 4]);
        }

        // ---- phase A: row max ----
        float mx = -3.0e38f;
        #pragma unroll
        for (int k = 0; k < 16; ++k) mx = fmaxf(mx, alphaS[r8][seg * 16 + k]);
        pred[r8][seg] = mx;
        __syncthreads();
        if (tid < ROWS) {
            float m = pred[tid][0];
            #pragma unroll
            for (int k = 1; k < 16; ++k) m = fmaxf(m, pred[tid][k]);
            rowm[tid] = m;
        }
        __syncthreads();
        // ---- phase A2: a16 = exp(alpha - m) in f16 ----
        {
            const float m = rowm[r8];
            #pragma unroll
            for (int k = 0; k < 16; ++k)
                aexp[r8][seg * 16 + k] = (_Float16)__expf(alphaS[r8][seg * 16 + k] - m);
        }
        __syncthreads();

        // ---- phase B: (16 x 256) @ (256 x 32) slice per wave via WMMA ----
        f8 acc0 = {};
        f8 acc1 = {};
        #pragma unroll
        for (int kc = 0; kc < 8; ++kc) {
            const int kb = kc * 32;
            // A fragment (16x32 f16): lane -> row mrow; K = kb + half*16 .. +15 contiguous
            const _Float16* ap = &aexp[mrow][kb + half * 16];
            h8 alo = *(const h8*)ap;
            h8 ahi = *(const h8*)(ap + 8);
            h16 afrag = __builtin_shufflevector(alo, ahi,
                0, 1, 2, 3, 4, 5, 6, 7, 8, 9, 10, 11, 12, 13, 14, 15);
            // B fragments (32x16 f16): lane -> col, K = kb + half*16 .. +15 contiguous
            const int kbb = kb + half * 16;
            const _Float16* bp0 = &expPT[wave * 32 + mrow][kbb];
            h8 b0lo = *(const h8*)bp0;
            h8 b0hi = *(const h8*)(bp0 + 8);
            h16 bfrag0 = __builtin_shufflevector(b0lo, b0hi,
                0, 1, 2, 3, 4, 5, 6, 7, 8, 9, 10, 11, 12, 13, 14, 15);
            acc0 = __builtin_amdgcn_wmma_f32_16x16x32_f16(
                false, afrag, false, bfrag0, (short)0, acc0, false, false);
            const _Float16* bp1 = &expPT[wave * 32 + 16 + mrow][kbb];
            h8 b1lo = *(const h8*)bp1;
            h8 b1hi = *(const h8*)(bp1 + 8);
            h16 bfrag1 = __builtin_shufflevector(b1lo, b1hi,
                0, 1, 2, 3, 4, 5, 6, 7, 8, 9, 10, 11, 12, 13, 14, 15);
            acc1 = __builtin_amdgcn_wmma_f32_16x16x32_f16(
                false, afrag, false, bfrag1, (short)0, acc1, false, false);
        }

        // ---- drain async emissions, sync, then phase C from LDS ----
        wait_async0();
        __syncthreads();

        const int ncol = wave * 32 + mrow;
        #pragma unroll
        for (int v = 0; v < 8; ++v) {
            const int mr = v + 8 * half;        // C/D layout: VGPR v, lane half
            alphaS[mr][ncol]      = rowm[mr] + __logf(acc0[v]) + lbuf[mr][ncol];
            alphaS[mr][ncol + 16] = rowm[mr] + __logf(acc1[v]) + lbuf[mr][ncol + 16];
        }
        __syncthreads();
    }

    // ---- log_Z = logsumexp_j(alpha_T + E) per batch row ----
    float mx = -3.0e38f;
    #pragma unroll
    for (int k = 0; k < 16; ++k)
        mx = fmaxf(mx, alphaS[r8][seg * 16 + k] + E[seg * 16 + k]);
    pred[r8][seg] = mx;
    __syncthreads();
    if (tid < ROWS) {
        float m = pred[tid][0];
        #pragma unroll
        for (int k = 1; k < 16; ++k) m = fmaxf(m, pred[tid][k]);
        rowm[tid] = m;
    }
    __syncthreads();
    float s = 0.0f;
    #pragma unroll
    for (int k = 0; k < 16; ++k)
        s += __expf(alphaS[r8][seg * 16 + k] + E[seg * 16 + k] - rowm[r8]);
    pred[r8][seg] = s;
    __syncthreads();
    if (tid < ROWS) {
        float acc = 0.0f;
        #pragma unroll
        for (int k = 0; k < 16; ++k) acc += pred[tid][k];
        logZ[b0 + tid] = rowm[tid] + __logf(acc);
    }
}

// ---------------------------------------------------------------------------
// Gold-path score: log_M[b] = sum_t emit + sum_t trans + S[y0] + E[yT-1]
// ---------------------------------------------------------------------------
__global__ __launch_bounds__(256)
void crf_gold_path(const float* __restrict__ logits,
                   const int*   __restrict__ labels,
                   const float* __restrict__ P,
                   const float* __restrict__ S,
                   const float* __restrict__ E,
                   float* __restrict__ logM)
{
    __shared__ float red[256];
    const int b   = blockIdx.x;
    const int tid = threadIdx.x;
    float s = 0.0f;
    for (int t = tid; t < TT; t += 256) {
        const int lab = labels[b * TT + t];
        s += logits[((size_t)b * TT + t) * VV + lab];
        if (t + 1 < TT) {
            const int labn = labels[b * TT + t + 1];
            s += P[lab * VV + labn];
        }
    }
    red[tid] = s;
    __syncthreads();
    for (int stride = 128; stride > 0; stride >>= 1) {
        if (tid < stride) red[tid] += red[tid + stride];
        __syncthreads();
    }
    if (tid == 0) {
        float tot = red[0];
        tot += S[labels[b * TT]] + E[labels[b * TT + TT - 1]];
        logM[b] = tot;
    }
}

// ---------------------------------------------------------------------------
// nll = mean_b(log_Z - log_M)
// ---------------------------------------------------------------------------
__global__ void crf_finalize(const float* __restrict__ logZ,
                             const float* __restrict__ logM,
                             float* __restrict__ out)
{
    if (threadIdx.x == 0) {
        float s = 0.0f;
        for (int b = 0; b < BB; ++b) s += logZ[b] - logM[b];
        out[0] = s / (float)BB;
    }
}

extern "C" void kernel_launch(void* const* d_in, const int* in_sizes, int n_in,
                              void* d_out, int out_size, void* d_ws, size_t ws_size,
                              hipStream_t stream) {
    const float* logits = (const float*)d_in[0];
    const int*   labels = (const int*)  d_in[1];
    const float* P      = (const float*)d_in[2];
    const float* S      = (const float*)d_in[3];
    const float* E      = (const float*)d_in[4];
    float* out  = (float*)d_out;
    float* logZ = (float*)d_ws;        // 64 floats
    float* logM = logZ + BB;           // 64 floats

    crf_forward_scan<<<BB / ROWS, 256, 0, stream>>>(logits, P, S, E, logZ);
    crf_gold_path<<<BB, 256, 0, stream>>>(logits, labels, P, S, E, logM);
    crf_finalize<<<1, 64, 0, stream>>>(logZ, logM, out);
}
